// TemplateAngleEmbedder_74672301408610
// MI455X (gfx1250) — matrix operand order repaired and checked by reading, not compiled, not run on hardware
//
#include <hip/hip_runtime.h>

typedef _Float16 v16h __attribute__((ext_vector_type(16)));
typedef float    v8f  __attribute__((ext_vector_type(8)));
typedef unsigned int v8u __attribute__((ext_vector_type(8)));

#define BN_TOT 65536     // 32 * 2048
#define NRES   2048
#define C_OUT  384
#define KPAD   64        // 57 padded to 64
#define FEAT_STRIDE 68   // padded LDS stride (halves) for s_feats rows
#define ACT_STRIDE  392  // padded LDS stride (halves) for s_act rows

// ---------------- constant tables (AlphaFold chi definitions) ----------------
__constant__ int c_chi_idx[21][4][4] = {
  /*ALA*/ {{0,0,0,0},{0,0,0,0},{0,0,0,0},{0,0,0,0}},
  /*ARG*/ {{0,1,3,5},{1,3,5,11},{3,5,11,23},{5,11,23,32}},
  /*ASN*/ {{0,1,3,5},{1,3,5,16},{0,0,0,0},{0,0,0,0}},
  /*ASP*/ {{0,1,3,5},{1,3,5,16},{0,0,0,0},{0,0,0,0}},
  /*CYS*/ {{0,1,3,10},{0,0,0,0},{0,0,0,0},{0,0,0,0}},
  /*GLN*/ {{0,1,3,5},{1,3,5,11},{3,5,11,26},{0,0,0,0}},
  /*GLU*/ {{0,1,3,5},{1,3,5,11},{3,5,11,26},{0,0,0,0}},
  /*GLY*/ {{0,0,0,0},{0,0,0,0},{0,0,0,0},{0,0,0,0}},
  /*HIS*/ {{0,1,3,5},{1,3,5,14},{0,0,0,0},{0,0,0,0}},
  /*ILE*/ {{0,1,3,6},{1,3,6,12},{0,0,0,0},{0,0,0,0}},
  /*LEU*/ {{0,1,3,5},{1,3,5,12},{0,0,0,0},{0,0,0,0}},
  /*LYS*/ {{0,1,3,5},{1,3,5,11},{3,5,11,19},{5,11,19,35}},
  /*MET*/ {{0,1,3,5},{1,3,5,18},{3,5,18,19},{0,0,0,0}},
  /*PHE*/ {{0,1,3,5},{1,3,5,12},{0,0,0,0},{0,0,0,0}},
  /*PRO*/ {{0,1,3,5},{1,3,5,11},{0,0,0,0},{0,0,0,0}},
  /*SER*/ {{0,1,3,8},{0,0,0,0},{0,0,0,0},{0,0,0,0}},
  /*THR*/ {{0,1,3,9},{0,0,0,0},{0,0,0,0},{0,0,0,0}},
  /*TRP*/ {{0,1,3,5},{1,3,5,12},{0,0,0,0},{0,0,0,0}},
  /*TYR*/ {{0,1,3,5},{1,3,5,12},{0,0,0,0},{0,0,0,0}},
  /*VAL*/ {{0,1,3,6},{0,0,0,0},{0,0,0,0},{0,0,0,0}},
  /*UNK*/ {{0,0,0,0},{0,0,0,0},{0,0,0,0},{0,0,0,0}}
};
__constant__ float c_chi_mask[21][4] = {
  {0,0,0,0},{1,1,1,1},{1,1,0,0},{1,1,0,0},{1,0,0,0},{1,1,1,0},{1,1,1,0},
  {0,0,0,0},{1,1,0,0},{1,1,0,0},{1,1,0,0},{1,1,1,1},{1,1,1,0},{1,1,0,0},
  {1,1,0,0},{1,0,0,0},{1,0,0,0},{1,1,0,0},{1,1,0,0},{1,0,0,0},{0,0,0,0}
};
__constant__ float c_chi_pi[21][4] = {
  {0,0,0,0},{0,0,0,0},{0,0,0,0},{0,1,0,0},{0,0,0,0},{0,0,0,0},{0,0,1,0},
  {0,0,0,0},{0,0,0,0},{0,0,0,0},{0,0,0,0},{0,0,0,0},{0,0,0,0},{0,1,0,0},
  {0,0,0,0},{0,0,0,0},{0,0,0,0},{0,0,0,0},{0,1,0,0},{0,0,0,0},{0,0,0,0}
};

// ---------------- small float3 helpers ----------------
struct F3 { float x, y, z; };
__device__ inline F3 f3sub(F3 a, F3 b){ return F3{a.x-b.x, a.y-b.y, a.z-b.z}; }
__device__ inline F3 f3scl(F3 a, float s){ return F3{a.x*s, a.y*s, a.z*s}; }
__device__ inline float f3dot(F3 a, F3 b){ return a.x*b.x + a.y*b.y + a.z*b.z; }
__device__ inline F3 f3cross(F3 a, F3 b){
  return F3{a.y*b.z - a.z*b.y, a.z*b.x - a.x*b.z, a.x*b.y - a.y*b.x};
}
__device__ inline F3 f3nrm(F3 v){
  float inv = 1.0f / sqrtf(f3dot(v, v) + 1e-8f);
  return f3scl(v, inv);
}
__device__ inline void dihedral(F3 p_xy, F3 p_negx, F3 origin, F3 p4,
                                float& s, float& c) {
  F3 e0  = f3nrm(f3sub(origin, p_negx));
  F3 e1u = f3sub(p_xy, origin);
  F3 e1  = f3nrm(f3sub(e1u, f3scl(e0, f3dot(e1u, e0))));
  F3 e2  = f3cross(e0, e1);
  F3 d   = f3sub(p4, origin);
  s = f3dot(e2, d);
  c = f3dot(e1, d);
  float inv = 1.0f / sqrtf(s*s + c*c + 1e-8f);
  s *= inv; c *= inv;
}

// Compute the 57 features (+7 zero pad) for residue i into dst (LDS, padded row)
__device__ inline void compute_features(int i, const int* __restrict__ aatype,
                                        const float* __restrict__ pos,
                                        const float* __restrict__ mask,
                                        _Float16* dst) {
  int n = i % NRES;
  int aa_orig = aatype[i];
  int aa = aa_orig < 20 ? aa_orig : 20;

  const float* P  = pos  + (size_t)i * 111;   // 37*3
  const float* Mk = mask + (size_t)i * 37;
  const float* Pp = P  - 111;
  const float* Mp = Mk - 37;
  bool hp = (n > 0);
  F3 zero{0.f, 0.f, 0.f};

  auto gp = [](const float* b, int a) { return F3{b[a*3], b[a*3+1], b[a*3+2]}; };

  F3 pN  = gp(P, 0), pCA = gp(P, 1), pC = gp(P, 2), pO = gp(P, 4);
  F3 ppCA = hp ? gp(Pp, 1) : zero;
  F3 ppC  = hp ? gp(Pp, 2) : zero;
  float mN = Mk[0], mCA = Mk[1], mC = Mk[2], mO = Mk[4];
  float mpCA = hp ? Mp[1] : 0.f;
  float mpC  = hp ? Mp[2] : 0.f;

  float sn[7], cs[7], mk[7], mir[7];

  dihedral(ppCA, ppC, pN, pCA, sn[0], cs[0]);      // pre_omega
  mk[0] = mpCA * mpC * mN * mCA; mir[0] = 1.f;
  dihedral(ppC, pN, pCA, pC, sn[1], cs[1]);        // phi
  mk[1] = mpC * mN * mCA * mC;   mir[1] = 1.f;
  dihedral(pN, pCA, pC, pO, sn[2], cs[2]);         // psi (sign-flipped)
  sn[2] = -sn[2]; cs[2] = -cs[2];
  mk[2] = mN * mCA * mC * mO;    mir[2] = 1.f;

  #pragma unroll
  for (int c = 0; c < 4; ++c) {
    int i0 = c_chi_idx[aa][c][0], i1 = c_chi_idx[aa][c][1];
    int i2 = c_chi_idx[aa][c][2], i3 = c_chi_idx[aa][c][3];
    dihedral(gp(P, i0), gp(P, i1), gp(P, i2), gp(P, i3), sn[3 + c], cs[3 + c]);
    mk[3 + c]  = c_chi_mask[aa][c] * Mk[i0] * Mk[i1] * Mk[i2] * Mk[i3];
    mir[3 + c] = 1.f - 2.f * c_chi_pi[aa][c];
  }

  #pragma unroll
  for (int j = 0; j < 22; ++j) dst[j] = (_Float16)((j == aa_orig) ? 1.f : 0.f);
  #pragma unroll
  for (int t = 0; t < 7; ++t) {
    dst[22 + 2*t] = (_Float16)sn[t];
    dst[23 + 2*t] = (_Float16)cs[t];
    dst[36 + 2*t] = (_Float16)(sn[t] * mir[t]);
    dst[37 + 2*t] = (_Float16)(cs[t] * mir[t]);
    dst[50 + t]   = (_Float16)mk[t];
  }
  #pragma unroll
  for (int j = 57; j < KPAD; ++j) dst[j] = (_Float16)0.f;
}

// ---------------- weight pack: f32 row-major -> f16 WMMA-B fragment layout ----
// dword index = ((kt*NT + nt)*32 + lane)*8 + v ; holds K = kt*32 + (lane>>4)*16 + 2v (+1)
__global__ void pack_weights(const float* __restrict__ src, unsigned int* __restrict__ dst,
                             int Ksrc, int Kpad, int N) {
  int id = blockIdx.x * blockDim.x + threadIdx.x;
  int total = (Kpad * N) >> 1;
  if (id >= total) return;
  int NT = N >> 4;
  int v    = id & 7;
  int lane = (id >> 3) & 31;
  int t2   = id >> 8;
  int nt   = t2 % NT;
  int kt   = t2 / NT;
  int k = kt * 32 + ((lane >> 4) << 4) + (v << 1);
  int n = nt * 16 + (lane & 15);
  float lo = (k     < Ksrc) ? src[(size_t)k * N + n]       : 0.0f;
  float hi = (k + 1 < Ksrc) ? src[(size_t)(k + 1) * N + n] : 0.0f;
  _Float16 lh = (_Float16)lo, hh = (_Float16)hi;
  unsigned int pl = __builtin_bit_cast(unsigned short, lh);
  unsigned int ph = __builtin_bit_cast(unsigned short, hh);
  dst[id] = pl | (ph << 16);
}

// A fragment from row-major f16 (global or LDS): lane m = lane%16,
// vgpr v holds K = (lane>>4)*16 + v*2 (+1)
__device__ inline v16h load_a_frag(const _Float16* A, int kstride_halves,
                                   int row, int k0, int lane) {
  const unsigned int* au =
      (const unsigned int*)(A + (size_t)row * kstride_halves + k0);
  int half = lane >> 4;
  v8u u;
  #pragma unroll
  for (int v = 0; v < 8; ++v)
    u[v] = au[(half << 3) + v];
  return __builtin_bit_cast(v16h, u);
}

// ---------------- fused kernel: features -> GEMM1(relu) -> GEMM2 ----------------
// One block = 128 output rows; activation tile lives entirely in LDS.
__global__ __launch_bounds__(256) void fused_kernel(
    const int* __restrict__ aatype, const float* __restrict__ pos,
    const float* __restrict__ mask,
    const unsigned int* __restrict__ w1p, const float* __restrict__ b1,
    const unsigned int* __restrict__ w2p, const float* __restrict__ b2,
    float* __restrict__ out) {
  __shared__ _Float16 s_feats[128 * FEAT_STRIDE];  // ~17.4 KB
  __shared__ _Float16 s_act  [128 * ACT_STRIDE];   // ~100.4 KB

  const int NT_TOT = 24;                 // 384/16
  int tid = threadIdx.x;
  size_t row0 = (size_t)blockIdx.x * 128;

  // ---- Phase A: per-row torsion features into LDS ----
  if (tid < 128) {
    compute_features((int)row0 + tid, aatype, pos, mask,
                     s_feats + tid * FEAT_STRIDE);
  }
  __syncthreads();

  int lane = tid & 31, wave = tid >> 5;
  int lm = lane & 15, half = lane >> 4;
  int lr0 = wave * 16;                   // this wave's 16-row strip (local)

  // ---- Phase B: act = relu(feats @ w1 + b1), kept in LDS ----
  #pragma unroll
  for (int c0 = 0; c0 < 3; ++c0) {       // 3 chunks of 8 N-tiles = 384 cols
    v8f acc[8] = {};
    #pragma unroll
    for (int kt = 0; kt < 2; ++kt) {
      v16h a = load_a_frag(s_feats, FEAT_STRIDE, lr0 + lm, kt * 32, lane);
      // Issue all 8 B-fragment loads first (one clause), then 8 WMMAs:
      // avoids a full s_wait_loadcnt 0x0 before every WMMA.
      v16h bf[8];
      #pragma unroll
      for (int t = 0; t < 8; ++t) {
        const v8u* bp = (const v8u*)w1p +
            (((kt * NT_TOT + c0 * 8 + t) << 5) + lane);
        bf[t] = __builtin_bit_cast(v16h, *bp);
      }
      #pragma unroll
      for (int t = 0; t < 8; ++t)
        acc[t] = __builtin_amdgcn_wmma_f32_16x16x32_f16(
            false, a, false, bf[t], (short)0, acc[t], false, false);
    }
    #pragma unroll
    for (int t = 0; t < 8; ++t) {
      int col = (c0 * 8 + t) * 16 + lm;
      float bb = b1[col];
      #pragma unroll
      for (int c = 0; c < 8; ++c) {
        float x = fmaxf(acc[t][c] + bb, 0.0f);
        s_act[(lr0 + c + 8 * half) * ACT_STRIDE + col] = (_Float16)x;
      }
    }
  }
  __syncthreads();

  // ---- Phase C: out = act @ w2 + b2 ----
  #pragma unroll
  for (int c0 = 0; c0 < 3; ++c0) {
    v8f acc[8] = {};
    #pragma unroll
    for (int kt = 0; kt < 12; ++kt) {    // K = 384
      v16h a = load_a_frag(s_act, ACT_STRIDE, lr0 + lm, kt * 32, lane);
      v16h bf[8];
      #pragma unroll
      for (int t = 0; t < 8; ++t) {
        const v8u* bp = (const v8u*)w2p +
            (((kt * NT_TOT + c0 * 8 + t) << 5) + lane);
        bf[t] = __builtin_bit_cast(v16h, *bp);
      }
      #pragma unroll
      for (int t = 0; t < 8; ++t)
        acc[t] = __builtin_amdgcn_wmma_f32_16x16x32_f16(
            false, a, false, bf[t], (short)0, acc[t], false, false);
    }
    #pragma unroll
    for (int t = 0; t < 8; ++t) {
      int col = (c0 * 8 + t) * 16 + lm;
      float bb = b2[col];
      #pragma unroll
      for (int c = 0; c < 8; ++c)
        out[(row0 + lr0 + c + 8 * half) * C_OUT + col] = acc[t][c] + bb;
    }
  }
}

// ---------------- launch ----------------
extern "C" void kernel_launch(void* const* d_in, const int* in_sizes, int n_in,
                              void* d_out, int out_size, void* d_ws, size_t ws_size,
                              hipStream_t stream) {
  const int*   aatype = (const int*)d_in[0];
  const float* pos    = (const float*)d_in[1];
  const float* mask   = (const float*)d_in[2];
  const float* w1     = (const float*)d_in[3];
  const float* b1     = (const float*)d_in[4];
  const float* w2     = (const float*)d_in[5];
  const float* b2     = (const float*)d_in[6];
  float* out = (float*)d_out;

  char* ws = (char*)d_ws;
  unsigned int* w1p = (unsigned int*)ws;                      // 48 KB
  unsigned int* w2p = (unsigned int*)(ws + (size_t)49152);    // 288 KB

  pack_weights<<<(12288 + 255) / 256, 256, 0, stream>>>(w1, w1p, 57, KPAD, C_OUT);
  pack_weights<<<(73728 + 255) / 256, 256, 0, stream>>>(w2, w2p, C_OUT, C_OUT, C_OUT);
  fused_kernel<<<BN_TOT / 128, 256, 0, stream>>>(aatype, pos, mask,
                                                 w1p, b1, w2p, b2, out);
}
